// FastVARAdaLNSelfAttn_35338990911760
// MI455X (gfx1250) — matrix operand order repaired and checked by reading, not compiled, not run on hardware
//
#include <hip/hip_runtime.h>
#include <hip/hip_bf16.h>

typedef __attribute__((ext_vector_type(16))) _Float16 v16h;
typedef __attribute__((ext_vector_type(8)))  _Float16 v8h;
typedef __attribute__((ext_vector_type(8)))  float    v8f;
typedef __attribute__((ext_vector_type(4)))  float    v4f;
typedef __attribute__((ext_vector_type(4)))  unsigned int u32x4;
typedef __attribute__((ext_vector_type(4)))  int      i32x4;
typedef __attribute__((ext_vector_type(8)))  int      i32x8;

// ---------------------------------------------------------------------------
// Fragment loaders per CDNA5 ISA 7.12.2 (wave32):
//  A 16x32 f16: lane(h=lane>>4, m=lane&15): v0-3 = K[8h..8h+7], v4-7 = K[16+8h..]
//  B 32x16 f16: lane(h, n=lane&15):         v0-3 = K[8h..8h+7], v4-7 = K[16+8h..]
// ---------------------------------------------------------------------------
__device__ __forceinline__ v16h frag_a(const _Float16* rowp, int hh) {
  v8h lo = *(const v8h*)(rowp + 8 * hh);
  v8h hi = *(const v8h*)(rowp + 16 + 8 * hh);
  return __builtin_shufflevector(lo, hi, 0,1,2,3,4,5,6,7,8,9,10,11,12,13,14,15);
}
__device__ __forceinline__ v16h frag_b(const _Float16* rowp, int hh) {
  v8h lo = *(const v8h*)(rowp + 8 * hh);
  v8h hi = *(const v8h*)(rowp + 16 + 8 * hh);
  return __builtin_shufflevector(lo, hi, 0,1,2,3,4,5,6,7,8,9,10,11,12,13,14,15);
}
__device__ __forceinline__ v8f wmma16(v16h a, v16h b, v8f c) {
  return __builtin_amdgcn_wmma_f32_16x16x32_f16(false, a, false, b, (short)0, c,
                                                false, false);
}

// ---------------------------------------------------------------------------
// DPP16 16-lane reductions (lanes 0-15 and 16-31 reduce independently):
// quad_perm xor1, xor2, row_half_mirror, row_mirror — pure VALU, no LDS.
// ---------------------------------------------------------------------------
template<int CTRL>
__device__ __forceinline__ float dppf(float v) {
  return __int_as_float(__builtin_amdgcn_update_dpp(
      __float_as_int(v), __float_as_int(v), CTRL, 0xf, 0xf, true));
}
__device__ __forceinline__ float redmax16(float m) {
  m = fmaxf(m, dppf<0x0B1>(m));   // quad_perm [1,0,3,2]
  m = fmaxf(m, dppf<0x04E>(m));   // quad_perm [2,3,0,1]
  m = fmaxf(m, dppf<0x141>(m));   // row_half_mirror
  m = fmaxf(m, dppf<0x140>(m));   // row_mirror
  return m;
}
__device__ __forceinline__ float redsum16(float m) {
  m += dppf<0x0B1>(m);
  m += dppf<0x04E>(m);
  m += dppf<0x141>(m);
  m += dppf<0x140>(m);
  return m;
}

// ---------------------------------------------------------------------------
// TDM: load a 64x64 f16 tile (contiguous rows of 64 halfs) into LDS with a
// hardware-padded stride of 72 halfs (row = 32 DWORDs -> pad_interval=4,
// pad 4 DWORDs -> pad_amount=3).  D# packing per CDNA5 ISA 8.3/8.4.
// Call from ONE wave only (TDM ignores EXEC; one issue per executing wave).
// ---------------------------------------------------------------------------
__device__ __forceinline__ void tdm_load_tile64x64(unsigned lds_off,
                                                   const _Float16* gptr) {
  const unsigned long long ga = (unsigned long long)gptr;
  u32x4 g0;
  g0[0] = 1u;                                   // count=1, user descriptor
  g0[1] = lds_off;                              // lds_addr (bytes)
  g0[2] = (unsigned)ga;                         // global_addr[31:0]
  g0[3] = (unsigned)(ga >> 32) | (2u << 30);    // global_addr[56:32] | type=2
  i32x8 g1;
  g1[0] = (int)0x07110000u;   // data_size=2B | pad_enable | interval=32dw | amt=4dw
  g1[1] = (int)(64u << 16);   // tensor_dim0 = 64 (bits 79:48 low half)
  g1[2] = (int)(64u << 16);   // tensor_dim0 hi=0 | tensor_dim1 = 64 low
  g1[3] = (int)(64u << 16);   // tensor_dim1 hi=0 | tile_dim0 = 64
  g1[4] = 64;                 // tile_dim1 = 64, tile_dim2 = 0
  g1[5] = 64;                 // tensor_dim0_stride = 64
  g1[6] = 0;
  g1[7] = 0;
  const i32x4 z4 = {0, 0, 0, 0};
#if defined(__clang_major__) && (__clang_major__ >= 23)
  const i32x8 z8 = {0, 0, 0, 0, 0, 0, 0, 0};
  __builtin_amdgcn_tensor_load_to_lds(g0, g1, z4, z4, z8, 0);
#else
  __builtin_amdgcn_tensor_load_to_lds(g0, g1, z4, z4, 0);
#endif
}

// ---------------------------------------------------------------------------
// GEMM staging helpers: load one 128x32 A slab chunk + B slab chunk to regs.
// ---------------------------------------------------------------------------
__device__ __forceinline__ void cvt16(const float* p, v8h& h0, v8h& h1) {
  v4f a0 = *(const v4f*)p,       a1 = *(const v4f*)(p + 4);
  v4f a2 = *(const v4f*)(p + 8), a3 = *(const v4f*)(p + 12);
  #pragma unroll
  for (int i = 0; i < 4; ++i) {
    h0[i] = (_Float16)a0[i]; h0[4 + i] = (_Float16)a1[i];
    h1[i] = (_Float16)a2[i]; h1[4 + i] = (_Float16)a3[i];
  }
}

// ---------------------------------------------------------------------------
// GEMM: out[M,N] = A[M,K] * W[N,K]^T + bias.  128x128 block, 8 waves (2x4),
// each wave 64x32 (4x2 accumulators).  Double-buffered LDS: global loads for
// tile i+1 issued before the WMMAs of tile i; one barrier per K-step.
// QKV mode: A = f32 x, bias = concat(q_bias, 0, v_bias), out = f16 qkv buffer.
// Proj mode: A = f16 attn-out, bias = proj_b, out = f32 d_out.
// ---------------------------------------------------------------------------
template<bool QKV>
__global__ __launch_bounds__(256)
void gemm_wmma(const float* __restrict__ Af, const _Float16* __restrict__ Ah,
               const float* __restrict__ W,
               const float* __restrict__ b0, const float* __restrict__ b2,
               _Float16* __restrict__ outH, float* __restrict__ outF,
               int M, int N, int K)
{
  __shared__ _Float16 As[2][128 * 40];   // stride 40 halfs (16B aligned rows)
  __shared__ _Float16 Bs[2][128 * 40];

  const int tid  = threadIdx.x;
  const int lane = tid & 31, wave = tid >> 5;
  const int hh = lane >> 4, mm = lane & 15;
  const int wm = wave >> 2, wn = wave & 3;
  const int mBase = blockIdx.y * 128, nBase = blockIdx.x * 128;
  const int srow = tid >> 1, scol = (tid & 1) * 16;   // staging: 2 thr/row x 16

  const float*    apF = QKV ? Af + (size_t)(mBase + srow) * K + scol : nullptr;
  const _Float16* apH = QKV ? nullptr : Ah + (size_t)(mBase + srow) * K + scol;
  const float*    bp  = W + (size_t)(nBase + srow) * K + scol;

  v8f acc[4][2] = {};
  const int nk = K >> 5;

  // ---- prologue: stage chunk 0 into buffer 0
  {
    v8h r0, r1, r2, r3;
    if (QKV) cvt16(apF, r0, r1);
    else { r0 = *(const v8h*)apH; r1 = *(const v8h*)(apH + 8); }
    cvt16(bp, r2, r3);
    *(v8h*)&As[0][srow * 40 + scol]     = r0;
    *(v8h*)&As[0][srow * 40 + scol + 8] = r1;
    *(v8h*)&Bs[0][srow * 40 + scol]     = r2;
    *(v8h*)&Bs[0][srow * 40 + scol + 8] = r3;
  }
  __syncthreads();

  for (int ki = 0; ki < nk; ++ki) {
    const int buf = ki & 1;
    const bool has_next = (ki + 1) < nk;
    v8h r0, r1, r2, r3;
    if (has_next) {                        // global loads for tile ki+1 first
      const int k0 = (ki + 1) << 5;
      if (QKV) cvt16(apF + k0, r0, r1);
      else { r0 = *(const v8h*)(apH + k0); r1 = *(const v8h*)(apH + k0 + 8); }
      cvt16(bp + k0, r2, r3);
      if (ki + 2 < nk) __builtin_prefetch(bp + k0 + 32, 0, 0);
    }

    v16h aF[4], bF[2];
    #pragma unroll
    for (int i = 0; i < 4; ++i)
      aF[i] = frag_a(&As[buf][(wm * 64 + i * 16 + mm) * 40], hh);
    #pragma unroll
    for (int j = 0; j < 2; ++j)
      bF[j] = frag_b(&Bs[buf][(wn * 32 + j * 16 + mm) * 40], hh);
    #pragma unroll
    for (int i = 0; i < 4; ++i)
      #pragma unroll
      for (int j = 0; j < 2; ++j)
        acc[i][j] = wmma16(aF[i], bF[j], acc[i][j]);

    if (has_next) {
      *(v8h*)&As[buf ^ 1][srow * 40 + scol]     = r0;
      *(v8h*)&As[buf ^ 1][srow * 40 + scol + 8] = r1;
      *(v8h*)&Bs[buf ^ 1][srow * 40 + scol]     = r2;
      *(v8h*)&Bs[buf ^ 1][srow * 40 + scol + 8] = r3;
      __syncthreads();
    }
  }

  // Epilogue. C layout: VGPR r -> M = r + 8*hh, lane&15 -> N.
  #pragma unroll
  for (int j = 0; j < 2; ++j) {
    const int gcol = nBase + wn * 32 + j * 16 + mm;
    float bj;
    if (QKV) bj = (gcol < 1024) ? b0[gcol]
                : ((gcol < 2048) ? 0.0f : b2[gcol - 2048]);
    else     bj = b0[gcol];
    #pragma unroll
    for (int i = 0; i < 4; ++i)
      #pragma unroll
      for (int r = 0; r < 8; ++r) {
        const int grow = mBase + wm * 64 + i * 16 + r + 8 * hh;
        const float v = acc[i][j][r] + bj;
        if (QKV) outH[(size_t)grow * N + gcol] = (_Float16)v;
        else     outF[(size_t)grow * N + gcol] = v;
      }
  }
}

// ---------------------------------------------------------------------------
// RoPE + scatter: qkv[B,L,3,H,D] f16 -> Q,K,V [B,H,L,D] f16.
// Pair (d, d+32), angle = pos * 10000^(-d/32). Q pre-scaled by 0.25/sqrt(D).
// ---------------------------------------------------------------------------
__global__ __launch_bounds__(256)
void rope_kernel(const _Float16* __restrict__ qkv, const int* __restrict__ pos,
                 _Float16* __restrict__ Q, _Float16* __restrict__ Kk,
                 _Float16* __restrict__ V)
{
  const int idx = blockIdx.x * 256 + threadIdx.x;   // B*L*H*32 = 2,097,152
  const int d = idx & 31;
  const int h = (idx >> 5) & 15;
  const int l = (idx >> 9) & 2047;
  const int b = idx >> 20;

  const size_t rbase = (size_t)(b * 2048 + l) * 3072;
  const size_t qi = rbase + h * 64 + d;
  const float q0 = (float)qkv[qi],        q1 = (float)qkv[qi + 32];
  const float k0 = (float)qkv[qi + 1024], k1 = (float)qkv[qi + 1024 + 32];
  const float v0 = (float)qkv[qi + 2048], v1 = (float)qkv[qi + 2048 + 32];

  const float p   = (float)pos[b * 2048 + l];
  const float inv = exp2f(-(float)d * 0.41524101186092f); // log2(1e4)/32
  const float ang = p * inv;
  const float c = cosf(ang), s = sinf(ang);
  const float scale = 0.25f / 8.0f;                 // 0.25 / sqrt(64)

  const size_t o = ((size_t)(b * 16 + h) * 2048 + l) * 64 + d;
  Q[o]       = (_Float16)((q0 * c - q1 * s) * scale);
  Q[o + 32]  = (_Float16)((q1 * c + q0 * s) * scale);
  Kk[o]      = (_Float16)(k0 * c - k1 * s);
  Kk[o + 32] = (_Float16)(k1 * c + k0 * s);
  V[o]       = (_Float16)v0;
  V[o + 32]  = (_Float16)v1;
}

// ---------------------------------------------------------------------------
// Flash attention: block = (128 q rows) x (b*h); 8 waves x 16 q rows each.
// K/V tiles DMA'd by the Tensor Data Mover, double-buffered: the TDM for tile
// kt+1 is issued before the WMMAs of tile kt and waited at the end, so the
// DMA runs under compute.  DPP16 row reductions (no LDS bpermute traffic).
// Per-wave exact causal tile skipping; mask VALU only on diagonal tiles.
// ---------------------------------------------------------------------------
__global__ __launch_bounds__(256)
void attn_kernel(const _Float16* __restrict__ Q, const _Float16* __restrict__ Kk,
                 const _Float16* __restrict__ V, _Float16* __restrict__ Aout)
{
  const int L = 2048;
  __shared__ _Float16 ks[2][64 * 72];     // [kRow][d], stride 72 (TDM-padded)
  __shared__ _Float16 vs[2][64 * 72];     // [vRow][d], raw V tile (TDM-padded)
  __shared__ _Float16 vts[64 * 72];       // [d][vRow] transposed (single buf)
  __shared__ _Float16 ps[8 * 16 * 72];    // per-wave P tile [16][64], stride 72

  const int tid  = threadIdx.x;
  const int lane = tid & 31, wave = tid >> 5;
  const int hh = lane >> 4, mm = lane & 15;
  const int bh = blockIdx.y;              // b*16 + h
  const int qt = blockIdx.x;              // q tile of 128
  const int qrow0 = qt * 128 + wave * 16;
  const unsigned ks_off[2] = { (unsigned)(size_t)&ks[0][0],
                               (unsigned)(size_t)&ks[1][0] };
  const unsigned vs_off[2] = { (unsigned)(size_t)&vs[0][0],
                               (unsigned)(size_t)&vs[1][0] };
  const _Float16* kBase = Kk + (size_t)bh * L * 64;
  const _Float16* vBase = V  + (size_t)bh * L * 64;

  // Resident Q fragments (already scaled): rows qrow0+mm, two K=32 chunks.
  const _Float16* qb = Q + ((size_t)bh * L + qrow0) * 64;
  v16h qF[2];
  #pragma unroll
  for (int kc = 0; kc < 2; ++kc)
    qF[kc] = frag_a(qb + (size_t)mm * 64 + 32 * kc, hh);

  v8f oA[4] = {};
  float mR[8], lR[8];
  #pragma unroll
  for (int r = 0; r < 8; ++r) { mR[r] = -1e30f; lR[r] = 0.0f; }

  const int nkt = 2 * (qt + 1);           // causal: only tiles <= block diagonal
  if (tid < 32) {                         // preload tile 0
    tdm_load_tile64x64(ks_off[0], kBase);
    tdm_load_tile64x64(vs_off[0], vBase);
    __builtin_amdgcn_s_wait_tensorcnt(0);
  }
  __syncthreads();

  for (int kt = 0; kt < nkt; ++kt) {
    const int kb = kt * 64;
    const int cur = kt & 1;
    {   // transpose V: vs[cur][row][d] -> vts[d][row], 256 thr cover 64x64
      const int row = tid >> 2, cb = (tid & 3) * 16;
      v8h va = *(const v8h*)&vs[cur][row * 72 + cb];
      v8h vb = *(const v8h*)&vs[cur][row * 72 + cb + 8];
      #pragma unroll
      for (int i = 0; i < 8; ++i) vts[(cb + i) * 72 + row]     = va[i];
      #pragma unroll
      for (int i = 0; i < 8; ++i) vts[(cb + 8 + i) * 72 + row] = vb[i];
    }
    if (tid < 32 && kt + 1 < nkt) {       // DMA next tile under compute
      tdm_load_tile64x64(ks_off[cur ^ 1], kBase + (size_t)(kb + 64) * 64);
      tdm_load_tile64x64(vs_off[cur ^ 1], vBase + (size_t)(kb + 64) * 64);
    }
    __syncthreads();                      // vts (and tile cur) visible

    if (kb <= qrow0 + 15) {               // wave-exact causal skip
      // S = q * k^T  (16 x 64), 4 n-tiles x 2 K-chunks
      v8f s[4] = {};
      #pragma unroll
      for (int nt = 0; nt < 4; ++nt)
        #pragma unroll
        for (int kc = 0; kc < 2; ++kc) {
          v16h bF = frag_b(&ks[cur][(nt * 16 + mm) * 72 + 32 * kc], hh);
          s[nt] = wmma16(qF[kc], bF, s[nt]);
        }

      // causal mask (only on diagonal-straddling tiles) + row max
      float mx[8];
      #pragma unroll
      for (int r = 0; r < 8; ++r) mx[r] = -1e30f;
      const bool needMask = (kb + 63 > qrow0);
      if (needMask) {
        #pragma unroll
        for (int nt = 0; nt < 4; ++nt) {
          const int col = kb + nt * 16 + mm;
          #pragma unroll
          for (int r = 0; r < 8; ++r) {
            const int rowg = qrow0 + r + 8 * hh;
            float sv = s[nt][r];
            if (col > rowg) sv = -1e30f;
            s[nt][r] = sv;
            mx[r] = fmaxf(mx[r], sv);
          }
        }
      } else {
        #pragma unroll
        for (int nt = 0; nt < 4; ++nt)
          #pragma unroll
          for (int r = 0; r < 8; ++r) mx[r] = fmaxf(mx[r], s[nt][r]);
      }
      float fac[8];
      #pragma unroll
      for (int r = 0; r < 8; ++r) {
        const float mn = fmaxf(mR[r], redmax16(mx[r]));
        fac[r] = __expf(mR[r] - mn);
        mR[r] = mn;
      }
      float rs[8] = {};
      #pragma unroll
      for (int nt = 0; nt < 4; ++nt)
        #pragma unroll
        for (int r = 0; r < 8; ++r) {
          const float pv = __expf(s[nt][r] - mR[r]);
          s[nt][r] = pv;
          rs[r] += pv;
        }
      #pragma unroll
      for (int r = 0; r < 8; ++r)
        lR[r] = lR[r] * fac[r] + redsum16(rs[r]);
      #pragma unroll
      for (int dt = 0; dt < 4; ++dt)
        #pragma unroll
        for (int r = 0; r < 8; ++r) oA[dt][r] *= fac[r];

      // P: C-layout regs -> per-wave LDS (row major f16) -> A-layout fragments
      const int pbase = wave * 16 * 72;
      #pragma unroll
      for (int nt = 0; nt < 4; ++nt)
        #pragma unroll
        for (int r = 0; r < 8; ++r)
          ps[pbase + (r + 8 * hh) * 72 + nt * 16 + mm] = (_Float16)s[nt][r];

      v16h pF[2];
      #pragma unroll
      for (int kc = 0; kc < 2; ++kc)
        pF[kc] = frag_a(&ps[pbase + mm * 72 + 32 * kc], hh);

      #pragma unroll
      for (int dt = 0; dt < 4; ++dt)
        #pragma unroll
        for (int kc = 0; kc < 2; ++kc) {
          v16h bV = frag_b(&vts[(dt * 16 + mm) * 72 + 32 * kc], hh);
          oA[dt] = wmma16(pF[kc], bV, oA[dt]);
        }
    }

    if (tid < 32 && kt + 1 < nkt)
      __builtin_amdgcn_s_wait_tensorcnt(0);   // next tile landed
    __syncthreads();                          // publish; all done reading cur
  }

  // normalize and write attn-out as f16 in [B, L, C] (C = h*64 + d)
  const int b = bh >> 4, h = bh & 15;
  #pragma unroll
  for (int dt = 0; dt < 4; ++dt)
    #pragma unroll
    for (int r = 0; r < 8; ++r) {
      const int l = qrow0 + r + 8 * hh;
      const float v = oA[dt][r] / lR[r];
      Aout[(size_t)(b * 2048 + l) * 1024 + h * 64 + dt * 16 + mm] = (_Float16)v;
    }
}

// ---------------------------------------------------------------------------
extern "C" void kernel_launch(void* const* d_in, const int* in_sizes, int n_in,
                              void* d_out, int out_size, void* d_ws, size_t ws_size,
                              hipStream_t stream) {
  (void)in_sizes; (void)n_in; (void)out_size; (void)ws_size;
  const float* x      = (const float*)d_in[0];
  // d_in[1] = attn_bias: full causal mask, reproduced analytically in-kernel
  const int*   pos    = (const int*)d_in[2];
  const float* qkv_w  = (const float*)d_in[3];
  const float* q_bias = (const float*)d_in[4];
  const float* v_bias = (const float*)d_in[5];
  const float* proj_w = (const float*)d_in[6];
  const float* proj_b = (const float*)d_in[7];
  float* out = (float*)d_out;

  char* ws = (char*)d_ws;
  _Float16* qkv = (_Float16*)ws;                          // 4096*3072 f16 = 24 MB
  _Float16* Q   = (_Float16*)(ws + 25165824);             // 8 MB each
  _Float16* Kk  = (_Float16*)(ws + 25165824 + 8388608);
  _Float16* V   = (_Float16*)(ws + 25165824 + 2 * 8388608);
  _Float16* AO  = (_Float16*)(ws + 25165824 + 3 * 8388608);

  gemm_wmma<true><<<dim3(24, 32), 256, 0, stream>>>(
      x, nullptr, qkv_w, q_bias, v_bias, qkv, nullptr, 4096, 3072, 1024);
  rope_kernel<<<8192, 256, 0, stream>>>(qkv, pos, Q, Kk, V);
  attn_kernel<<<dim3(16, 32), 256, 0, stream>>>(Q, Kk, V, AO);
  gemm_wmma<false><<<dim3(8, 32), 256, 0, stream>>>(
      nullptr, AO, proj_w, proj_b, nullptr, nullptr, out, 4096, 1024, 1024);
}
